// SignatureTransform_300647710813
// MI455X (gfx1250) — compile-verified, matrix operand order checked by reading, not run on hardware
//
#include <hip/hip_runtime.h>
#include <hip/hip_bf16.h>

// ---------------------------------------------------------------------------
// Path signature, depth 4, path (B=32, L=1024, d=8) f32.
// out = [L1 (32x8) | L2 (32x64) | L3 (32x512) | L4 (32x4096)] flat f32.
//
// Two-level (chunked) Chen scan:
//   K1 chunk scan : 32 chunks x 32 steps per batch (serial depth 32).
//   K2..K4 WMMA   : per-chunk level-4 via three outer-product GEMMs (K=32,
//                   one v_wmma_f32_16x16x32_f16 per 16x16 tile), each into
//                   its OWN buffer (pure stores, no RMW, static masks):
//                     L4a = Sum_t (S3+x3/4) (x) x    (512x8, x^4/24 folded)
//                     L4b = Sum_t S2 (x) x2          (64x64)
//                     L4d = Sum_t S1 (x) x3          (8x512)
//   K5 combine    : per batch, 31 serial Chen merges; P3 staged into LDS via
//                   double-buffered GLOBAL_LOAD_ASYNC_TO_LDS_B32 (ASYNCcnt).
// Fallback (ws too small): pure-LDS serial kernel incl. level 4.
// ---------------------------------------------------------------------------

typedef __attribute__((ext_vector_type(16))) _Float16 v16h;
typedef __attribute__((ext_vector_type(8)))  _Float16 v8h;
typedef __attribute__((ext_vector_type(8)))  float    v8f;

#define NB   32      // batches
#define NT   1024    // increments (1023 real + 1 zero pad, pow2)
#define ND   8       // path dimension
#define NC   32      // chunks
#define CT   32      // steps per chunk (== WMMA K)
#define L1OFF 0
#define L2OFF (NB*8)                       // 256
#define L3OFF (L2OFF + NB*64)              // 2304
#define L4OFF (L3OFF + NB*512)             // 18688

// --- gfx1250 async global->LDS helpers (ASYNCcnt) ---------------------------
__device__ __forceinline__ void async_g2lds_b32(void* lds_dst, const void* gsrc) {
  const unsigned laddr = (unsigned)(unsigned long long)lds_dst;  // LDS byte addr
  asm volatile("global_load_async_to_lds_b32 %0, %1, off"
               :: "v"(laddr), "v"(gsrc) : "memory");
}
__device__ __forceinline__ void wait_async_le1() {
  asm volatile("s_wait_asynccnt 0x1" ::: "memory");
}
__device__ __forceinline__ void wait_async_le0() {
  asm volatile("s_wait_asynccnt 0x0" ::: "memory");
}

// ---------------------------------------------------------------------------
// K1: chunk-local scan. grid = (NC, NB), 512 threads.
// Streams are K-major: stream[(b*W + row)*NT + t], W padded to mult of 16.
// ---------------------------------------------------------------------------
__global__ __launch_bounds__(512)
void sig_chunk_scan(const float* __restrict__ path,
                    _Float16* __restrict__ a3s, _Float16* __restrict__ s2s,
                    _Float16* __restrict__ s1s, _Float16* __restrict__ xs,
                    _Float16* __restrict__ x2s, _Float16* __restrict__ x3s,
                    float* __restrict__ P1, float* __restrict__ P2,
                    float* __restrict__ P3) {
  __shared__ float S1[8], S2[64], S3[512], X[8], X2[64];
  const int tid = threadIdx.x;
  const int c   = blockIdx.x;
  const int b   = blockIdx.y;
  if (tid < 8)  S1[tid] = 0.f;
  if (tid < 64) S2[tid] = 0.f;
  S3[tid] = 0.f;
  __syncthreads();

  const float* P = path + (size_t)b * NT * ND;
  const size_t r512 = (size_t)(b * 512 + tid) * NT;
  const size_t r64  = (size_t)(b * 64 + tid) * NT;
  const size_t r16  = (size_t)(b * 16 + tid) * NT;   // padded width-16 streams

  for (int s = 0; s < CT; ++s) {
    const int t = c * CT + s;
    if (tid < 8) X[tid] = (t < NT - 1) ? (P[(t + 1) * ND + tid] - P[t * ND + tid]) : 0.f;
    __syncthreads();
    if (tid < 64) X2[tid] = 0.5f * X[tid >> 3] * X[tid & 7];
    __syncthreads();
    const float v3 = (1.f / 3.f) * X2[tid >> 3] * X[tid & 7];   // x^3/6 element

    // Emit streams (chunk-local values BEFORE this step's update).
    a3s[r512 + t] = (_Float16)(S3[tid] + 0.25f * v3);
    x3s[r512 + t] = (_Float16)v3;
    if (tid < 64) { s2s[r64 + t] = (_Float16)S2[tid]; x2s[r64 + t] = (_Float16)X2[tid]; }
    if (tid < 16) {                                   // rows 8..15 = zero padding
      s1s[r16 + t] = (_Float16)(tid < 8 ? S1[tid] : 0.f);
      xs[r16 + t]  = (_Float16)(tid < 8 ? X[tid]  : 0.f);
    }

    // Chen step for levels 1..3.
    const float n3 = S3[tid] + S2[tid >> 3] * X[tid & 7]
                             + S1[tid >> 6] * X2[tid & 63] + v3;
    const float n2 = (tid < 64) ? (S2[tid] + S1[tid >> 3] * X[tid & 7] + X2[tid]) : 0.f;
    const float n1 = (tid < 8)  ? (S1[tid] + X[tid]) : 0.f;
    __syncthreads();
    S3[tid] = n3;
    if (tid < 64) S2[tid] = n2;
    if (tid < 8)  S1[tid] = n1;
    __syncthreads();
  }

  const size_t pb = (size_t)(b * NC + c);
  if (tid < 8)  P1[pb * 8 + tid]  = S1[tid];
  if (tid < 64) P2[pb * 64 + tid] = S2[tid];
  P3[pb * 512 + tid] = S3[tid];
}

// ---------------------------------------------------------------------------
// K2..K4: per-chunk level-4 GEMM, compile-time shapes. K = CT = 32 -> one
// wmma per 16x16 tile, 8 waves / 256-thread block, pure stores (own buffer).
// M,N true dims; MP,NP padded stream widths (mult of 16, zero-padded rows).
// Canonical store index j = mr*N + n covers 0..4095 exactly per reshape.
// ---------------------------------------------------------------------------
template <int M, int N, int MP, int NP>
__global__ __launch_bounds__(256)
void sig_l4c_gemm(const _Float16* __restrict__ As, const _Float16* __restrict__ Bs,
                  float* __restrict__ L4) {
  constexpr int ntiles = (N + 15) / 16;
  constexpr int mtiles = (M + 15) / 16;
  constexpr int tiles  = mtiles * ntiles;
  const int b   = blockIdx.y;
  const int gt  = blockIdx.x * (blockDim.x >> 5) + (threadIdx.x >> 5);
  const int c   = gt / tiles;              // chunk
  const int tix = gt % tiles;
  const int mt  = tix / ntiles;
  const int nt  = tix % ntiles;
  const int l   = threadIdx.x & 31;
  const int l15 = l & 15;
  const int h   = l >> 4;                  // half-wave K sub-run select
  const int m   = mt * 16 + l15;           // < MP (padded)
  const int n   = nt * 16 + l15;           // < NP (padded)

  const _Float16* Arow = As + (size_t)(b * MP + m) * NT + c * CT;
  const _Float16* Brow = Bs + (size_t)(b * NP + n) * NT + c * CT;

  // Lane fragment: K = 8h..8h+7 and K = 16+8h..16+8h+7 (two b128 loads each).
  const v8h alo = *(const v8h*)(Arow + 8 * h);
  const v8h ahi = *(const v8h*)(Arow + 16 + 8 * h);
  const v8h blo = *(const v8h*)(Brow + 8 * h);
  const v8h bhi = *(const v8h*)(Brow + 16 + 8 * h);
  v16h a, bm;
#pragma unroll
  for (int i = 0; i < 8; ++i) { a[i] = alo[i]; a[i + 8] = ahi[i]; }
#pragma unroll
  for (int i = 0; i < 8; ++i) { bm[i] = blo[i]; bm[i + 8] = bhi[i]; }
  v8f cacc = {};
  cacc = __builtin_amdgcn_wmma_f32_16x16x32_f16(
      /*neg_a=*/false, a, /*neg_b=*/false, bm,
      /*c_mod=*/(short)0, cacc, /*reuse_a=*/false, /*reuse_b=*/false);

  float* outc = L4 + (size_t)(b * NC + c) * 4096;
#pragma unroll
  for (int v = 0; v < 8; ++v) {
    const int mr = mt * 16 + v + 8 * h;
    if ((M % 16 == 0 || mr < M) && (N % 16 == 0 || n < N))   // static when padded
      outc[(size_t)mr * N + n] = cacc[v];
  }
}

// ---------------------------------------------------------------------------
// K5: per-batch serial Chen combine of NC chunk signatures. grid = NB, 512 thr.
// P3 staged via double-buffered async global->LDS; L4 partials summed on read.
// ---------------------------------------------------------------------------
__global__ __launch_bounds__(512)
void sig_combine(const float* __restrict__ P1, const float* __restrict__ P2,
                 const float* __restrict__ P3, const float* __restrict__ L4a,
                 const float* __restrict__ L4b, const float* __restrict__ L4d,
                 float* __restrict__ out) {
  __shared__ float A1[8], A2[64], A3[512], p1[8], p2[64], p3buf[2][512];
  const int tid = threadIdx.x;
  const int b   = blockIdx.x;
  float A4[8] = {0.f, 0.f, 0.f, 0.f, 0.f, 0.f, 0.f, 0.f};
  if (tid < 8)  A1[tid] = 0.f;
  if (tid < 64) A2[tid] = 0.f;
  A3[tid] = 0.f;

  // Preload chunk 0's P3 into buffer 0 (async, tracked by ASYNCcnt).
  async_g2lds_b32(&p3buf[0][tid], P3 + (size_t)(b * NC) * 512 + tid);
  __syncthreads();

  for (int c = 0; c < NC; ++c) {
    const size_t pb = (size_t)(b * NC + c);
    const bool more = (c + 1 < NC);
    if (more)   // prefetch next chunk's P3 into the alternate buffer
      async_g2lds_b32(&p3buf[(c + 1) & 1][tid], P3 + (pb + 1) * 512 + tid);
    if (tid < 8)  p1[tid] = P1[pb * 8 + tid];
    if (tid < 64) p2[tid] = P2[pb * 64 + tid];
    if (more) wait_async_le1(); else wait_async_le0();   // oldest (chunk c) done
    __syncthreads();

    const float* p3c = p3buf[c & 1];
    const float* la = L4a + pb * 4096;
    const float* lb = L4b + pb * 4096;
    const float* ld = L4d + pb * 4096;
    float n4[8];
#pragma unroll
    for (int u = 0; u < 8; ++u) {
      const int j = tid * 8 + u;
      n4[u] = A4[u] + la[j] + lb[j] + ld[j]
            + A3[j >> 3] * p1[j & 7] + A2[j >> 6] * p2[j & 63]
            + A1[j >> 9] * p3c[j & 511];
    }
    const float n3 = A3[tid] + p3c[tid] + A2[tid >> 3] * p1[tid & 7]
                   + A1[tid >> 6] * p2[tid & 63];
    const float n2 = (tid < 64) ? (A2[tid] + p2[tid] + A1[tid >> 3] * p1[tid & 7]) : 0.f;
    const float n1 = (tid < 8)  ? (A1[tid] + p1[tid]) : 0.f;
    __syncthreads();
#pragma unroll
    for (int u = 0; u < 8; ++u) A4[u] = n4[u];
    A3[tid] = n3;
    if (tid < 64) A2[tid] = n2;
    if (tid < 8)  A1[tid] = n1;
  }
  __syncthreads();

  if (tid < 8)  out[L1OFF + b * 8 + tid]  = A1[tid];
  if (tid < 64) out[L2OFF + b * 64 + tid] = A2[tid];
  out[L3OFF + (size_t)b * 512 + tid] = A3[tid];
#pragma unroll
  for (int u = 0; u < 8; ++u)
    out[L4OFF + (size_t)b * 4096 + tid * 8 + u] = A4[u];
}

// ---------------------------------------------------------------------------
// Fallback: fully serial scan including level 4 in LDS.
// ---------------------------------------------------------------------------
__global__ __launch_bounds__(512)
void sig_scan_full(const float* __restrict__ path, float* __restrict__ out) {
  __shared__ float S1[8], S2[64], S3[512], S4[4096], X[8], X2[64], X3[512];
  const int tid = threadIdx.x;
  const int b   = blockIdx.x;
  if (tid < 8)  S1[tid] = 0.f;
  if (tid < 64) S2[tid] = 0.f;
  S3[tid] = 0.f;
#pragma unroll
  for (int u = 0; u < 8; ++u) S4[tid * 8 + u] = 0.f;
  __syncthreads();

  const float* P = path + (size_t)b * NT * ND;
  for (int t = 0; t < NT; ++t) {
    if (tid < 8) X[tid] = (t < NT - 1) ? (P[(t + 1) * ND + tid] - P[t * ND + tid]) : 0.f;
    __syncthreads();
    if (tid < 64) X2[tid] = 0.5f * X[tid >> 3] * X[tid & 7];
    __syncthreads();
    X3[tid] = (1.f / 3.f) * X2[tid >> 3] * X[tid & 7];
    __syncthreads();

    float n4[8];
#pragma unroll
    for (int u = 0; u < 8; ++u) {
      const int j = tid * 8 + u;
      n4[u] = S4[j] + S3[j >> 3] * X[j & 7] + S2[j >> 6] * X2[j & 63]
            + S1[j >> 9] * X3[j & 511] + 0.25f * X3[j >> 3] * X[j & 7];
    }
    const float n3 = S3[tid] + S2[tid >> 3] * X[tid & 7]
                             + S1[tid >> 6] * X2[tid & 63] + X3[tid];
    const float n2 = (tid < 64) ? (S2[tid] + S1[tid >> 3] * X[tid & 7] + X2[tid]) : 0.f;
    const float n1 = (tid < 8)  ? (S1[tid] + X[tid]) : 0.f;
    __syncthreads();
#pragma unroll
    for (int u = 0; u < 8; ++u) S4[tid * 8 + u] = n4[u];
    S3[tid] = n3;
    if (tid < 64) S2[tid] = n2;
    if (tid < 8)  S1[tid] = n1;
    __syncthreads();
  }

  if (tid < 8)  out[L1OFF + b * 8 + tid]  = S1[tid];
  if (tid < 64) out[L2OFF + b * 64 + tid] = S2[tid];
  out[L3OFF + (size_t)b * 512 + tid] = S3[tid];
#pragma unroll
  for (int u = 0; u < 8; ++u)
    out[L4OFF + (size_t)b * 4096 + tid * 8 + u] = S4[tid * 8 + u];
}

// ---------------------------------------------------------------------------
extern "C" void kernel_launch(void* const* d_in, const int* in_sizes, int n_in,
                              void* d_out, int out_size, void* d_ws, size_t ws_size,
                              hipStream_t stream) {
  const float* path = (const float*)d_in[0];   // (32, 1024, 8) f32; d_in[1] = depth (4)
  float* out = (float*)d_out;

  // Workspace: f16 streams (padded widths) + f32 chunk partials + 3 L4 bufs.
  const size_t H512 = (size_t)NB * 512 * NT;
  const size_t H64  = (size_t)NB * 64 * NT;
  const size_t H16  = (size_t)NB * 16 * NT;
  const size_t halfs = 2 * H512 + 2 * H64 + 2 * H16;
  const size_t F1 = (size_t)NB * NC * 8;
  const size_t F2 = (size_t)NB * NC * 64;
  const size_t F3 = (size_t)NB * NC * 512;
  const size_t F4 = (size_t)NB * NC * 4096;
  const size_t need = halfs * sizeof(_Float16)
                    + (F1 + F2 + F3 + 3 * F4) * sizeof(float);

  if (ws_size >= need) {
    _Float16* w   = (_Float16*)d_ws;
    _Float16* a3s = w;  w += H512;
    _Float16* x3s = w;  w += H512;
    _Float16* s2s = w;  w += H64;
    _Float16* x2s = w;  w += H64;
    _Float16* s1s = w;  w += H16;
    _Float16* xs  = w;  w += H16;
    float* f   = (float*)w;
    float* P1  = f;  f += F1;
    float* P2  = f;  f += F2;
    float* P3  = f;  f += F3;
    float* L4a = f;  f += F4;
    float* L4b = f;  f += F4;
    float* L4d = f;

    sig_chunk_scan<<<dim3(NC, NB), 512, 0, stream>>>(path, a3s, s2s, s1s, xs,
                                                     x2s, x3s, P1, P2, P3);

    // L4a = Sum_t (S3 + x3/4) (x) x : 512x8, 32 tiles/chunk
    sig_l4c_gemm<512, 8, 512, 16>
        <<<dim3(32 * NC / 8, NB), 256, 0, stream>>>(a3s, xs, L4a);
    // L4b = Sum_t S2 (x) x2         : 64x64, 16 tiles/chunk
    sig_l4c_gemm<64, 64, 64, 64>
        <<<dim3(16 * NC / 8, NB), 256, 0, stream>>>(s2s, x2s, L4b);
    // L4d = Sum_t S1 (x) x3         : 8x512, 32 tiles/chunk
    sig_l4c_gemm<8, 512, 16, 512>
        <<<dim3(32 * NC / 8, NB), 256, 0, stream>>>(s1s, x3s, L4d);

    sig_combine<<<NB, 512, 0, stream>>>(P1, P2, P3, L4a, L4b, L4d, out);
  } else {
    sig_scan_full<<<NB, 512, 0, stream>>>(path, out);
  }
}
